// DynamicConv_39711267619164
// MI455X (gfx1250) — matrix operand order repaired and checked by reading, not compiled, not run on hardware
//
#include <hip/hip_runtime.h>
#include <hip/hip_bf16.h>
#include <stdint.h>

// Problem constants (reference: N=32, C=384, H=W=56, K=3, G=32)
#define NB    32
#define C_CH  384
#define NGRP  32
#define CPG   12          // C / G
#define HH    56
#define WW    56
#define HWP   (HH*WW)     // 3136
#define EPS   1e-5f

#define PG    49          // p groups of 64 (3136/64)
#define KSTEPS 12         // 384 / 32
#define BPAD  392         // padded LDS row stride (halves): 784B -> conflict-free ds_load_b128

typedef __bf16 v8bf  __attribute__((ext_vector_type(8)));
typedef __bf16 v16bf __attribute__((ext_vector_type(16)));
typedef float  v8f   __attribute__((ext_vector_type(8)));
typedef int    v4i   __attribute__((ext_vector_type(4)));

union F16 { v16bf v; v8bf h[2]; };

typedef __attribute__((address_space(1))) v4i as1_v4i;
typedef __attribute__((address_space(3))) v4i as3_v4i;

__device__ __forceinline__ unsigned short f2bf(float f) {
    unsigned int u = __float_as_uint(f);
    unsigned int r = u + 0x7FFFu + ((u >> 16) & 1u);   // round-to-nearest-even
    return (unsigned short)(r >> 16);
}

// ---- gfx1250 async global->LDS copy (ASYNCcnt), builtin if present else inline asm
__device__ __forceinline__ void async_copy_b128(const void* g, unsigned lds_off) {
#if __has_builtin(__builtin_amdgcn_global_load_async_to_lds_b128)
    __builtin_amdgcn_global_load_async_to_lds_b128(
        (as1_v4i*)(uintptr_t)g, (as3_v4i*)(uintptr_t)lds_off, 0, 0);
#else
    unsigned long long ga = (unsigned long long)(uintptr_t)g;
    asm volatile("global_load_async_to_lds_b128 %0, %1, off"
                 :: "v"(lds_off), "v"(ga) : "memory");
#endif
}

__device__ __forceinline__ void wait_async0() {
#if __has_builtin(__builtin_amdgcn_s_wait_asynccnt)
    __builtin_amdgcn_s_wait_asynccnt(0);
#else
    asm volatile("s_wait_asynccnt 0x0" ::: "memory");
#endif
}

// ---------------------------------------------------------------- stats zero
__global__ void zero_stats_kernel(float* stats) {
    int i = blockIdx.x * 256 + threadIdx.x;
    if (i < NB * NGRP * 2) stats[i] = 0.f;
}

// ---------------------------------------------------------------- global avg pool
__global__ void pool_kernel(const float* __restrict__ x, float* __restrict__ pooled) {
    int nc = blockIdx.x;                       // n*C + c
    const float4* xr = (const float4*)(x + (size_t)nc * HWP);
    float s = 0.f;
    for (int i = threadIdx.x; i < HWP / 4; i += 256) {
        float4 v = xr[i];
        s += v.x + v.y + v.z + v.w;
    }
    __shared__ float red[256];
    red[threadIdx.x] = s;
    __syncthreads();
    for (int off = 128; off > 0; off >>= 1) {
        if (threadIdx.x < off) red[threadIdx.x] += red[threadIdx.x + off];
        __syncthreads();
    }
    if (threadIdx.x == 0) pooled[nc] = red[0] * (1.f / (float)HWP);
}

// ---------------------------------------------------------------- kernel-gen MLP
__global__ void mlp_kernel(const float* __restrict__ pooled,
                           const float* __restrict__ w1, const float* __restrict__ b1,
                           const float* __restrict__ w2, const float* __restrict__ b2,
                           float* __restrict__ kern) {
    int n = blockIdx.x, j = threadIdx.x;       // 384 threads
    __shared__ float pin[C_CH], hbuf[C_CH];
    pin[j] = pooled[n * C_CH + j];
    __syncthreads();
    const float* wr = w1 + (size_t)j * C_CH;
    float acc = b1[j];
    for (int i = 0; i < C_CH; ++i) acc += pin[i] * wr[i];
    hbuf[j] = fmaxf(acc, 0.f);
    __syncthreads();
    for (int r = 0; r < 9; ++r) {
        int o = j * 9 + r;                     // (c, tap) row-major == reshape(N,C,3,3)
        const float* w2r = w2 + (size_t)o * C_CH;
        float a = b2[o];
        for (int i = 0; i < C_CH; ++i) a += hbuf[i] * w2r[i];
        kern[(size_t)n * C_CH * 9 + o] = a;
    }
}

// ---------------------------------------------------------------- proj_w -> bf16
__global__ void cvtw_kernel(const float* __restrict__ w, unsigned short* __restrict__ wbf) {
    int i = blockIdx.x * 256 + threadIdx.x;
    if (i < C_CH * C_CH) wbf[i] = f2bf(w[i]);
}

// ---------------------------------------------------------------- depthwise 3x3 -> y (channel-last bf16)
#define TPAD 392
__global__ void dwconv_kernel(const float* __restrict__ x,
                              const float* __restrict__ kern,
                              unsigned short* __restrict__ y) {
    int b = blockIdx.x;
    int n = b / (HH * 2);
    int rem = b % (HH * 2);
    int h = rem >> 1;
    int chunk = rem & 1;
    int w0 = chunk * 32;
    int wlen = chunk ? (WW - 32) : 32;

    int wl = threadIdx.x & 31;
    int cl = threadIdx.x >> 5;                 // 0..7
    __shared__ unsigned short tile[32 * TPAD]; // [w][c], padded vs bank conflicts

    for (int co = 0; co < C_CH / 8; ++co) {
        int c = co * 8 + cl;
        const float* kp = kern + ((size_t)n * C_CH + c) * 9;
        float k9[9];
        #pragma unroll
        for (int t = 0; t < 9; ++t) k9[t] = kp[t];
        if (wl < wlen) {
            int w = w0 + wl;
            const float* xc = x + ((size_t)(n * C_CH + c)) * HWP;
            float acc = 0.f;
            #pragma unroll
            for (int dh = -1; dh <= 1; ++dh) {
                int hh = h + dh;
                if ((unsigned)hh >= HH) continue;
                #pragma unroll
                for (int dw = -1; dw <= 1; ++dw) {
                    int ww2 = w + dw;
                    if ((unsigned)ww2 >= WW) continue;
                    acc += xc[hh * WW + ww2] * k9[(dh + 1) * 3 + (dw + 1)];
                }
            }
            tile[wl * TPAD + c] = f2bf(acc);
        }
    }
    __syncthreads();
    unsigned int* yd = (unsigned int*)(y + ((size_t)n * HWP + (h * WW + w0)) * C_CH);
    int totald = (wlen * C_CH) >> 1;
    for (int i = threadIdx.x; i < totald; i += 256) {
        int e = i * 2;
        int wl2 = e / C_CH;
        int c2  = e % C_CH;
        yd[i] = *(const unsigned int*)&tile[wl2 * TPAD + c2];
    }
}

// ---------------------------------------------------------------- WMMA GEMM + fused GN stats
// Block = (n, o-chunk of 128, p-group of 64). The 64x384 bf16 B slab is async-copied
// to LDS once (global_load_async_to_lds_b128 + s_wait_asynccnt), shared by 8 waves;
// each wave owns one 16-row o-tile, A double-buffered from global (L2-hot).
__global__ void gemm_kernel(const unsigned short* __restrict__ wbf,
                            const unsigned short* __restrict__ y,
                            const float* __restrict__ proj_b,
                            float* __restrict__ out,
                            float* __restrict__ stats) {
    __shared__ __attribute__((aligned(16))) unsigned short bslab[64 * BPAD]; // ~49 KB
    __shared__ float s1[NGRP], s2[NGRP];

    int b  = blockIdx.x;
    int n  = b / (3 * PG);
    int r  = b % (3 * PG);
    int oc = r / PG;                  // 0..2 (o-chunk of 8 tiles)
    int pg = r % PG;
    int p0 = pg * 64;

    for (int i = threadIdx.x; i < NGRP; i += 256) { s1[i] = 0.f; s2[i] = 0.f; }

    // Async-stage B slab: y[n][p0..p0+64)[0..384), contiguous in memory; rows padded in LDS.
    const unsigned short* ybase = y + ((size_t)n * HWP + p0) * C_CH;
    unsigned lds_base = (unsigned)(uintptr_t)(&bslab[0]);
    #pragma unroll
    for (int it = 0; it < 12; ++it) {                     // 3072 chunks of 16B
        int cid = it * 256 + threadIdx.x;
        int row = cid / 48;
        int col = (cid % 48) * 8;
        async_copy_b128(ybase + row * C_CH + col, lds_base + (row * BPAD + col) * 2u);
    }
    wait_async0();
    __syncthreads();

    int wid  = threadIdx.x >> 5;
    int lane = threadIdx.x & 31;
    int o0 = (oc * 8 + wid) * 16;
    bool lolane = lane < 16;
    int  m = lane & 15;

    // A fragment source (ISA 16-bit A 16x32 layout): row o0+m,
    // lanes 0-15: K {c0..c0+7, c0+16..c0+23}; lanes 16-31: +8.
    const unsigned short* wrow = wbf + (size_t)(o0 + m) * C_CH + (lolane ? 0 : 8);

    // B fragment source in LDS: column p_local = t*16+m, K {c0..c0+7, c0+16..c0+23} + (lo?0:8).
    const unsigned short* brow[4];
    #pragma unroll
    for (int t = 0; t < 4; ++t)
        brow[t] = &bslab[(t * 16 + m) * BPAD + (lolane ? 0 : 8)];

    v8f acc[4] = {};
    F16 a_cur, a_nxt;
    a_cur.h[0] = *(const v8bf*)(wrow);
    a_cur.h[1] = *(const v8bf*)(wrow + 16);

    #pragma unroll
    for (int k = 0; k < KSTEPS; ++k) {
        int c0 = k * 32;
        if (k + 1 < KSTEPS) {                   // double-buffered A prefetch
            __builtin_prefetch(wrow + c0 + 64, 0, 1);
            a_nxt.h[0] = *(const v8bf*)(wrow + c0 + 32);
            a_nxt.h[1] = *(const v8bf*)(wrow + c0 + 48);
        }
        // Materialize all four B fragments first (distinct registers -> the 8
        // ds_load_b128 issue back-to-back and WMMAs don't each wait dscnt==0).
        F16 bfr[4];
        #pragma unroll
        for (int t = 0; t < 4; ++t) {
            bfr[t].h[0] = *(const v8bf*)(brow[t] + c0);
            bfr[t].h[1] = *(const v8bf*)(brow[t] + c0 + 16);
        }
        #pragma unroll
        for (int t = 0; t < 4; ++t) {
            acc[t] = __builtin_amdgcn_wmma_f32_16x16x32_bf16(
                false, a_cur.v, false, bfr[t].v, (short)0, acc[t], false, false);
        }
        a_cur = a_nxt;
    }

    // Epilogue: bias, store fp32, fused GroupNorm partial sums.
    #pragma unroll
    for (int j = 0; j < 8; ++j) {
        int o = o0 + j + (lolane ? 0 : 8);     // C/D layout: VGPR j -> M = j (+8 hi half)
        float bo = proj_b[o];
        float vs = 0.f, vq = 0.f;
        float* orow = out + ((size_t)n * C_CH + o) * HWP + p0 + m;
        #pragma unroll
        for (int t = 0; t < 4; ++t) {
            float v = acc[t][j] + bo;
            orow[t * 16] = v;
            vs += v;
            vq += v * v;
        }
        for (int msk = 8; msk >= 1; msk >>= 1) {   // reduce within 16-lane half
            vs += __shfl_xor(vs, msk, 32);
            vq += __shfl_xor(vq, msk, 32);
        }
        if (m == 0) {
            int g = o / CPG;
            atomicAdd(&s1[g], vs);
            atomicAdd(&s2[g], vq);
        }
    }
    __syncthreads();
    for (int i = threadIdx.x; i < NGRP; i += 256) {
        atomicAdd(&stats[(n * NGRP + i) * 2 + 0], s1[i]);
        atomicAdd(&stats[(n * NGRP + i) * 2 + 1], s2[i]);
    }
}

// ---------------------------------------------------------------- GroupNorm finalize (in place, float4)
__global__ void gn_finalize_kernel(float* __restrict__ out,
                                   const float* __restrict__ stats,
                                   const float* __restrict__ gamma,
                                   const float* __restrict__ beta) {
    size_t t = (size_t)blockIdx.x * 256 + threadIdx.x;     // one float4 each
    const size_t P4 = HWP / 4;
    size_t o = (t / P4) % C_CH;
    size_t n = t / (P4 * C_CH);
    if (n >= NB) return;
    int g = (int)(o / CPG);
    float s  = stats[(n * NGRP + g) * 2 + 0];
    float sq = stats[(n * NGRP + g) * 2 + 1];
    const float cnt = (float)(CPG * HWP);
    float mu  = s / cnt;
    float var = sq / cnt - mu * mu;
    float inv = rsqrtf(var + EPS);
    float ga = gamma[o], be = beta[o];
    float4* ptr = (float4*)out + t;
    float4 v = *ptr;
    v.x = (v.x - mu) * inv * ga + be;
    v.y = (v.y - mu) * inv * ga + be;
    v.z = (v.z - mu) * inv * ga + be;
    v.w = (v.w - mu) * inv * ga + be;
    *ptr = v;
}

// ---------------------------------------------------------------- host launcher
extern "C" void kernel_launch(void* const* d_in, const int* in_sizes, int n_in,
                              void* d_out, int out_size, void* d_ws, size_t ws_size,
                              hipStream_t stream) {
    const float* x      = (const float*)d_in[0];
    const float* w1     = (const float*)d_in[1];
    const float* b1     = (const float*)d_in[2];
    const float* w2     = (const float*)d_in[3];
    const float* b2     = (const float*)d_in[4];
    const float* proj_w = (const float*)d_in[5];
    const float* proj_b = (const float*)d_in[6];
    const float* gamma  = (const float*)d_in[7];
    const float* beta   = (const float*)d_in[8];
    float* out = (float*)d_out;

    // Workspace carve-up (all offsets 256B aligned); total ~78 MB.
    char* w = (char*)d_ws;
    float* pooled = (float*)w;          w += (size_t)NB * C_CH * 4;
    float* kern   = (float*)w;          w += (size_t)NB * C_CH * 9 * 4;
    float* stats  = (float*)w;          w += (size_t)NB * NGRP * 2 * 4;
    unsigned short* wbf = (unsigned short*)w; w += (size_t)C_CH * C_CH * 2;
    unsigned short* y   = (unsigned short*)w;              // 73.5 MB

    zero_stats_kernel<<<(NB * NGRP * 2 + 255) / 256, 256, 0, stream>>>(stats);
    pool_kernel<<<NB * C_CH, 256, 0, stream>>>(x, pooled);
    mlp_kernel<<<NB, C_CH, 0, stream>>>(pooled, w1, b1, w2, b2, kern);
    cvtw_kernel<<<(C_CH * C_CH + 255) / 256, 256, 0, stream>>>(proj_w, wbf);
    dwconv_kernel<<<NB * HH * 2, 256, 0, stream>>>(x, kern, y);
    gemm_kernel<<<NB * 3 * PG, 256, 0, stream>>>(wbf, y, proj_b, out, stats);
    gn_finalize_kernel<<<(NB * C_CH * (HWP / 4) + 255) / 256, 256, 0, stream>>>(out, stats, gamma, beta);
}
